// RnnModel_8297876816296
// MI455X (gfx1250) — compile-verified, matrix operand order checked by reading, not run on hardware
//
#include <hip/hip_runtime.h>
#include <hip/hip_bf16.h>
#include <stdint.h>

// ---------------------------------------------------------------------------
// Bi-LSTM + combiner LSTM + dense head for MI455X (gfx1250, wave32, WMMA).
// All GEMMs: bf16 A/B, f32 accumulate via v_wmma_f32_16x16x32_bf16.
// Recurrences are batch-partitioned: each WG owns 16 batch rows and runs the
// full 32-step scan internally (no grid sync). x_t staging is double-buffered
// and uses CDNA5 async global->LDS loads when the toolchain exposes them.
// ---------------------------------------------------------------------------

typedef __bf16 bf16_t;
typedef __attribute__((ext_vector_type(16))) __bf16 v16bf;
typedef __attribute__((ext_vector_type(8)))  float  v8f;
typedef int v4i32 __attribute__((vector_size(16)));

constexpr int kS = 32;
constexpr int kB = 4096;
constexpr int kE = 64;     // per-direction hidden
constexpr int kH2 = 128;   // combiner hidden (2E)
constexpr int kV = 32000;

#if __has_builtin(__builtin_amdgcn_global_load_async_to_lds_b128) && \
    __has_builtin(__builtin_amdgcn_s_wait_asynccnt)
#define USE_ASYNC_LDS 1
#endif

// 16-byte global->LDS copy (async-to-LDS on CDNA5 if available).
__device__ __forceinline__ void copy16_to_lds(const void* g, void* l) {
#ifdef USE_ASYNC_LDS
    __builtin_amdgcn_global_load_async_to_lds_b128(
        (__attribute__((address_space(1))) v4i32*)g,
        (__attribute__((address_space(3))) v4i32*)l, 0, 0);
#else
    *(uint4*)l = *(const uint4*)g;
#endif
}

__device__ __forceinline__ void wait_async_stage() {
#ifdef USE_ASYNC_LDS
    __builtin_amdgcn_s_wait_asynccnt(0);
#endif
}

// slot-pair 'a' (0..7), half 'hi' (0/1) -> K offset of the low element.
__device__ __forceinline__ int frag_k0(int a, int hi) {
    return ((a & 4) ? (2 * a + 8) : (2 * a)) + 8 * hi;
}

// Load one 16x32 bf16 WMMA operand fragment from a row-major matrix.
// For A: row = m (absolute). For B: row = n (absolute) over W stored (N,K)
// row-major, which *is* B = W^T in the k-striped lane layout.
__device__ __forceinline__ v16bf load_frag(const bf16_t* p, int ld, int row,
                                           int hi, int kbase) {
    union { uint32_t u[8]; v16bf v; } r;
    const uint32_t* p32 = (const uint32_t*)p;
    const int base = row * ld + kbase;
#pragma unroll
    for (int a = 0; a < 8; ++a) r.u[a] = p32[(base + frag_k0(a, hi)) >> 1];
    return r.v;
}

__device__ __forceinline__ v8f wmma_bf16(v16bf a, v16bf b, v8f c) {
    return __builtin_amdgcn_wmma_f32_16x16x32_bf16(false, a, false, b,
                                                   (short)0, c, false, false);
}

// Branch-free transcendentals on the v_exp_f32 / v_rcp_f32 TRANS units.
__device__ __forceinline__ float sig_fast(float x) {
    return __builtin_amdgcn_rcpf(
        1.0f + __builtin_amdgcn_exp2f(-1.4426950408889634f * x));
}
__device__ __forceinline__ float tanh_fast(float x) {
    const float e = __builtin_amdgcn_exp2f(-2.8853900817779268f * fabsf(x));
    const float r = (1.0f - e) * __builtin_amdgcn_rcpf(1.0f + e);
    return __builtin_copysignf(r, x);
}

// ---------------------------------------------------------------------------
// K0 helpers: f32->bf16 convert, bias fuse
// ---------------------------------------------------------------------------
__global__ void cvt_bf16_kernel(const float* __restrict__ src,
                                bf16_t* __restrict__ dst, int n) {
    int i = blockIdx.x * 256 + threadIdx.x;
    if (i < n) dst[i] = (bf16_t)src[i];
}

__global__ void bias_sum_kernel(const float* __restrict__ a,
                                const float* __restrict__ b,
                                float* __restrict__ o, int n) {
    int i = blockIdx.x * 256 + threadIdx.x;
    if (i < n) o[i] = a[i] + b[i];
}

// ---------------------------------------------------------------------------
// K1: forward+backward LSTM (hidden 64). grid (B/16, 2), block 128.
// Writes Hcat[t][b][dir*64 + hid] (bf16); backward in production order
// (exactly what the combiner consumes).
// ---------------------------------------------------------------------------
__global__ __launch_bounds__(128) void lstm_dir_kernel(
    const int* __restrict__ xtok,
    const float* __restrict__ h0f, const float* __restrict__ c0f,
    const float* __restrict__ h0b, const float* __restrict__ c0b,
    const bf16_t* __restrict__ emb_bf,
    const bf16_t* __restrict__ Wih_f, const bf16_t* __restrict__ Whh_f,
    const bf16_t* __restrict__ Wih_b, const bf16_t* __restrict__ Whh_b,
    const float* __restrict__ bsum_f, const float* __restrict__ bsum_b,
    bf16_t* __restrict__ Hcat) {
    const int dir = blockIdx.y;
    const int rowbase = blockIdx.x * 16;
    const int tid = threadIdx.x;

    const bf16_t* Wih = dir ? Wih_b : Wih_f;
    const bf16_t* Whh = dir ? Whh_b : Whh_f;
    const float* bsum = dir ? bsum_b : bsum_f;
    const float* h0 = dir ? h0b : h0f;
    const float* c0 = dir ? c0b : c0f;

    __shared__ __attribute__((aligned(16))) bf16_t xa[2][16][72];  // x_t, 2-buf
    __shared__ __attribute__((aligned(16))) bf16_t ha[16][72];     // h_t
    __shared__ float gates[16][256];
    __shared__ int toks[kS][16];

    // Preload all tokens this WG will ever need (removes the per-step
    // dependent index-load from the serial critical path).
    for (int q = tid; q < kS * 16; q += 128) {
        const int t = q >> 4, i = q & 15;
        toks[t][i] = xtok[t * kB + rowbase + i];
    }

    const int erow = tid >> 3;       // elementwise row 0..15
    const int ecol0 = tid & 7;       // hid = ecol0 + 8*p
    float creg[8];
#pragma unroll
    for (int p = 0; p < 8; ++p) {
        const int hid = ecol0 + 8 * p;
        creg[p] = c0[(rowbase + erow) * kE + hid];
        ha[erow][hid] = (bf16_t)h0[(rowbase + erow) * kE + hid];
    }
    __syncthreads();

    const int lane = tid & 31;
    const int wv = tid >> 5;
    const int lrow = lane & 15;
    const int hi = lane >> 4;
    const int si = tid >> 3, sc8 = tid & 7;   // staging role

    auto stage = [&](int step, int bufsel) {
        const int tt = dir ? (kS - 1 - step) : step;
        const int tok = toks[tt][si];
        copy16_to_lds(emb_bf + (size_t)tok * kE + sc8 * 8,
                      &xa[bufsel][si][sc8 * 8]);
    };

    stage(0, 0);
    int buf = 0;

    for (int t = 0; t < kS; ++t) {
        wait_async_stage();
        if (t + 1 < kS) stage(t + 1, buf ^ 1);   // overlap with this step
        __syncthreads();

#pragma unroll
        for (int pnl = 0; pnl < 4; ++pnl) {
            const int nsub = pnl * 64 + wv * 16;
            const float bv = bsum[nsub + lrow];
            v8f acc;
#pragma unroll
            for (int r = 0; r < 8; ++r) acc[r] = bv;
#pragma unroll
            for (int kc = 0; kc < 2; ++kc) {   // x @ Wih^T
                v16bf af = load_frag(&xa[buf][0][0], 72, lrow, hi, kc * 32);
                v16bf bf = load_frag(Wih, kE, nsub + lrow, hi, kc * 32);
                acc = wmma_bf16(af, bf, acc);
            }
#pragma unroll
            for (int kc = 0; kc < 2; ++kc) {   // h @ Whh^T
                v16bf af = load_frag(&ha[0][0], 72, lrow, hi, kc * 32);
                v16bf bf = load_frag(Whh, kE, nsub + lrow, hi, kc * 32);
                acc = wmma_bf16(af, bf, acc);
            }
#pragma unroll
            for (int r = 0; r < 8; ++r)
                gates[r + 8 * hi][nsub + lrow] = acc[r];
        }
        __syncthreads();

#pragma unroll
        for (int p = 0; p < 8; ++p) {
            const int hid = ecol0 + 8 * p;
            const float ig = gates[erow][hid];
            const float fg = gates[erow][kE + hid];
            const float gg = gates[erow][2 * kE + hid];
            const float og = gates[erow][3 * kE + hid];
            const float cn = sig_fast(fg) * creg[p] + sig_fast(ig) * tanh_fast(gg);
            const float hn = sig_fast(og) * tanh_fast(cn);
            creg[p] = cn;
            const bf16_t hb = (bf16_t)hn;
            ha[erow][hid] = hb;
            Hcat[(size_t)t * kB * kH2 + (size_t)(rowbase + erow) * kH2 +
                 dir * kE + hid] = hb;
        }
        __syncthreads();
        buf ^= 1;
    }
}

// ---------------------------------------------------------------------------
// K2: combiner LSTM (hidden 128, input 128 = [hf|hb]). grid (B/16), block 128.
// Only the final h is kept (bf16) for the dense head.
// ---------------------------------------------------------------------------
__global__ __launch_bounds__(128) void lstm_comb_kernel(
    const float* __restrict__ h0c, const float* __restrict__ c0c,
    const bf16_t* __restrict__ Hcat,
    const bf16_t* __restrict__ Wih_c, const bf16_t* __restrict__ Whh_c,
    const float* __restrict__ bsum_c, bf16_t* __restrict__ hLbf) {
    const int rowbase = blockIdx.x * 16;
    const int tid = threadIdx.x;

    __shared__ __attribute__((aligned(16))) bf16_t xa[2][16][136];
    __shared__ __attribute__((aligned(16))) bf16_t ha[16][136];
    __shared__ float gates[16][512];   // 32 KB

    const int erow = tid >> 3;
    const int ecol0 = tid & 7;
    float creg[16];
#pragma unroll
    for (int p = 0; p < 16; ++p) {
        const int hid = ecol0 + 8 * p;
        creg[p] = c0c[(rowbase + erow) * kH2 + hid];
        ha[erow][hid] = (bf16_t)h0c[(rowbase + erow) * kH2 + hid];
    }
    __syncthreads();

    const int lane = tid & 31;
    const int wv = tid >> 5;
    const int lrow = lane & 15;
    const int hi = lane >> 4;
    const int si = tid >> 3, sc8 = tid & 7;

    auto stage = [&](int step, int bufsel) {
        const bf16_t* src = Hcat + (size_t)step * kB * kH2 +
                            (size_t)(rowbase + si) * kH2 + sc8 * 16;
        copy16_to_lds(src, &xa[bufsel][si][sc8 * 16]);
        copy16_to_lds(src + 8, &xa[bufsel][si][sc8 * 16 + 8]);
    };

    stage(0, 0);
    int buf = 0;

    for (int t = 0; t < kS; ++t) {
        wait_async_stage();
        if (t + 1 < kS) stage(t + 1, buf ^ 1);
        __syncthreads();

#pragma unroll
        for (int pnl = 0; pnl < 8; ++pnl) {
            const int nsub = pnl * 64 + wv * 16;
            const float bv = bsum_c[nsub + lrow];
            v8f acc;
#pragma unroll
            for (int r = 0; r < 8; ++r) acc[r] = bv;
#pragma unroll
            for (int kc = 0; kc < 4; ++kc) {   // x @ Wih_c^T (K=128)
                v16bf af = load_frag(&xa[buf][0][0], 136, lrow, hi, kc * 32);
                v16bf bf = load_frag(Wih_c, kH2, nsub + lrow, hi, kc * 32);
                acc = wmma_bf16(af, bf, acc);
            }
#pragma unroll
            for (int kc = 0; kc < 4; ++kc) {   // h @ Whh_c^T (K=128)
                v16bf af = load_frag(&ha[0][0], 136, lrow, hi, kc * 32);
                v16bf bf = load_frag(Whh_c, kH2, nsub + lrow, hi, kc * 32);
                acc = wmma_bf16(af, bf, acc);
            }
#pragma unroll
            for (int r = 0; r < 8; ++r)
                gates[r + 8 * hi][nsub + lrow] = acc[r];
        }
        __syncthreads();

#pragma unroll
        for (int p = 0; p < 16; ++p) {
            const int hid = ecol0 + 8 * p;
            const float ig = gates[erow][hid];
            const float fg = gates[erow][kH2 + hid];
            const float gg = gates[erow][2 * kH2 + hid];
            const float og = gates[erow][3 * kH2 + hid];
            const float cn = sig_fast(fg) * creg[p] + sig_fast(ig) * tanh_fast(gg);
            const float hn = sig_fast(og) * tanh_fast(cn);
            creg[p] = cn;
            const bf16_t hb = (bf16_t)hn;
            ha[erow][hid] = hb;
            if (t == kS - 1)
                hLbf[(size_t)(rowbase + erow) * kH2 + hid] = hb;
        }
        __syncthreads();
        buf ^= 1;
    }
}

// ---------------------------------------------------------------------------
// K3: dense head  out = hL @ Wd^T + bd   (4096 x 32000, f32 out)
// grid (N/64, M/64), block 128; each wave does a 16x64 strip, K=128.
// Output is streamed with non-temporal stores (written once, never reread).
// ---------------------------------------------------------------------------
__global__ __launch_bounds__(128) void dense_kernel(
    const bf16_t* __restrict__ hLbf, const bf16_t* __restrict__ Wd_bf,
    const float* __restrict__ bd, float* __restrict__ out) {
    const int nbase = blockIdx.x * 64;
    const int wv = threadIdx.x >> 5;
    const int mbase = blockIdx.y * 64 + wv * 16;
    const int lane = threadIdx.x & 31;
    const int lrow = lane & 15;
    const int hi = lane >> 4;

    __builtin_prefetch(Wd_bf + (size_t)nbase * kH2, 0, 1);

    v8f acc[4];
#pragma unroll
    for (int nt = 0; nt < 4; ++nt) {
        const float bv = bd[nbase + nt * 16 + lrow];
#pragma unroll
        for (int r = 0; r < 8; ++r) acc[nt][r] = bv;
    }
#pragma unroll
    for (int kc = 0; kc < 4; ++kc) {
        v16bf af = load_frag(hLbf, kH2, mbase + lrow, hi, kc * 32);
#pragma unroll
        for (int nt = 0; nt < 4; ++nt) {
            v16bf bf = load_frag(Wd_bf, kH2, nbase + nt * 16 + lrow, hi, kc * 32);
            acc[nt] = wmma_bf16(af, bf, acc[nt]);
        }
    }
#pragma unroll
    for (int nt = 0; nt < 4; ++nt)
#pragma unroll
        for (int r = 0; r < 8; ++r)
            __builtin_nontemporal_store(
                acc[nt][r],
                &out[(size_t)(mbase + r + 8 * hi) * kV + nbase + nt * 16 + lrow]);
}

// ---------------------------------------------------------------------------
extern "C" void kernel_launch(void* const* d_in, const int* in_sizes, int n_in,
                              void* d_out, int out_size, void* d_ws,
                              size_t ws_size, hipStream_t stream) {
    const int*   x     = (const int*)d_in[0];
    const float* h0f   = (const float*)d_in[1];
    const float* c0f   = (const float*)d_in[2];
    const float* h0b   = (const float*)d_in[3];
    const float* c0b   = (const float*)d_in[4];
    const float* h0c   = (const float*)d_in[5];
    const float* c0c   = (const float*)d_in[6];
    const float* emb   = (const float*)d_in[7];
    const float* Wih_f = (const float*)d_in[8];
    const float* Whh_f = (const float*)d_in[9];
    const float* bih_f = (const float*)d_in[10];
    const float* bhh_f = (const float*)d_in[11];
    const float* Wih_b = (const float*)d_in[12];
    const float* Whh_b = (const float*)d_in[13];
    const float* bih_b = (const float*)d_in[14];
    const float* bhh_b = (const float*)d_in[15];
    const float* Wih_c = (const float*)d_in[16];
    const float* Whh_c = (const float*)d_in[17];
    const float* bih_c = (const float*)d_in[18];
    const float* bhh_c = (const float*)d_in[19];
    const float* Wd    = (const float*)d_in[20];
    const float* bd    = (const float*)d_in[21];
    float* out = (float*)d_out;

    uint8_t* w = (uint8_t*)d_ws;
    size_t off = 0;
    auto alloc = [&](size_t bytes) -> void* {
        void* p = w + off;
        off += (bytes + 255) & ~(size_t)255;
        return p;
    };
    bf16_t* emb_bf   = (bf16_t*)alloc((size_t)kV * kE * 2);
    bf16_t* Wih_f_bf = (bf16_t*)alloc((size_t)4 * kE * kE * 2);
    bf16_t* Whh_f_bf = (bf16_t*)alloc((size_t)4 * kE * kE * 2);
    bf16_t* Wih_b_bf = (bf16_t*)alloc((size_t)4 * kE * kE * 2);
    bf16_t* Whh_b_bf = (bf16_t*)alloc((size_t)4 * kE * kE * 2);
    bf16_t* Wih_c_bf = (bf16_t*)alloc((size_t)4 * kH2 * kH2 * 2);
    bf16_t* Whh_c_bf = (bf16_t*)alloc((size_t)4 * kH2 * kH2 * 2);
    bf16_t* Wd_bf    = (bf16_t*)alloc((size_t)kV * kH2 * 2);
    float*  bsum_f   = (float*)alloc(4 * kE * 4);
    float*  bsum_b   = (float*)alloc(4 * kE * 4);
    float*  bsum_c   = (float*)alloc(4 * kH2 * 4);
    bf16_t* Hcat     = (bf16_t*)alloc((size_t)kS * kB * kH2 * 2);
    bf16_t* hLbf     = (bf16_t*)alloc((size_t)kB * kH2 * 2);
    (void)ws_size; (void)n_in; (void)in_sizes; (void)out_size;

    auto cvt = [&](const float* s, bf16_t* d, int n) {
        cvt_bf16_kernel<<<(n + 255) / 256, 256, 0, stream>>>(s, d, n);
    };
    cvt(emb, emb_bf, kV * kE);
    cvt(Wih_f, Wih_f_bf, 4 * kE * kE);
    cvt(Whh_f, Whh_f_bf, 4 * kE * kE);
    cvt(Wih_b, Wih_b_bf, 4 * kE * kE);
    cvt(Whh_b, Whh_b_bf, 4 * kE * kE);
    cvt(Wih_c, Wih_c_bf, 4 * kH2 * kH2);
    cvt(Whh_c, Whh_c_bf, 4 * kH2 * kH2);
    cvt(Wd, Wd_bf, kV * kH2);
    bias_sum_kernel<<<1, 256, 0, stream>>>(bih_f, bhh_f, bsum_f, 4 * kE);
    bias_sum_kernel<<<1, 256, 0, stream>>>(bih_b, bhh_b, bsum_b, 4 * kE);
    bias_sum_kernel<<<2, 256, 0, stream>>>(bih_c, bhh_c, bsum_c, 4 * kH2);

    lstm_dir_kernel<<<dim3(kB / 16, 2), 128, 0, stream>>>(
        x, h0f, c0f, h0b, c0b, emb_bf, Wih_f_bf, Whh_f_bf, Wih_b_bf, Whh_b_bf,
        bsum_f, bsum_b, Hcat);

    lstm_comb_kernel<<<dim3(kB / 16), 128, 0, stream>>>(
        h0c, c0c, Hcat, Wih_c_bf, Whh_c_bf, bsum_c, hLbf);

    dense_kernel<<<dim3(kV / 64, kB / 64), 128, 0, stream>>>(hLbf, Wd_bf, bd,
                                                             out);
}
